// BertForTABSAJoint_CRF_75282186764997
// MI455X (gfx1250) — compile-verified, matrix operand order checked by reading, not run on hardware
//
#include <hip/hip_runtime.h>

// ---------------------------------------------------------------------------
// MI455X (gfx1250) BERT+CRF forward, v2.
//  - Weights pre-converted per layer to bf16 "pairs along K" layout [K/2,N] u32.
//  - Activations produced directly in bf16 by LN/GELU/attention epilogues.
//  - GEMM tile staging = pure byte copy -> CDNA5 async global->LDS B128 path
//    (global_load_async_to_lds_b128 + s_wait_asynccnt), zero inner-loop VALU
//    conversion. WMMA: v_wmma_f32_16x16x32_bf16, 8 waves/block, 128x64 tiles.
// ---------------------------------------------------------------------------

#define USE_ASYNC_LDS 1

typedef __attribute__((ext_vector_type(16))) __bf16 v16bf;
typedef __attribute__((ext_vector_type(2)))  __bf16 bf16x2;
typedef __attribute__((ext_vector_type(8)))  float  v8f;

union BFrag { v16bf v; unsigned int u[8]; };

__device__ __forceinline__ unsigned int pack_bf16(float a, float b) {
  bf16x2 p = { (__bf16)a, (__bf16)b };          // -> v_cvt_pk_bf16_f32 if available
  return __builtin_bit_cast(unsigned int, p);
}
__device__ __forceinline__ unsigned short bf16_1(float a) {
  return __builtin_bit_cast(unsigned short, (__bf16)a);
}

// ISA 7.12.2: 16-bit A 16x32 — VGPR v holds K pair starting at this k0.
__device__ __forceinline__ int kmap(int v, int lh) {
  return ((v < 4) ? (2 * v) : (16 + 2 * (v - 4))) + 8 * lh;
}

__device__ __forceinline__ float gelu_tanh(float x) {
  float x3 = x * x * x;
  float t  = tanhf(0.7978845608028654f * (x + 0.044715f * x3));
  return 0.5f * x * (1.0f + t);
}

__device__ __forceinline__ unsigned int lds_off(const void* p) {
  return (unsigned int)(size_t)p;   // flat->LDS: low 32 bits are the LDS offset
}

// ---------------------------------------------------------------------------
// Weight converter: f32 [K,N] -> u32 [K/2,N], each u32 = bf16(W[2kp][n]) |
// bf16(W[2kp+1][n])<<16  (exact B-fragment pairing along K).
// ---------------------------------------------------------------------------
__global__ void convert_w_kernel(const float* __restrict__ W,
                                 unsigned int* __restrict__ Wp,
                                 int halfK, int N)
{
  int idx = blockIdx.x * blockDim.x + threadIdx.x;
  if (idx >= halfK * N) return;
  int kp = idx / N, n = idx - kp * N;
  Wp[idx] = pack_bf16(W[(long)(2 * kp) * N + n], W[(long)(2 * kp + 1) * N + n]);
}

// ---------------------------------------------------------------------------
// WMMA GEMM: C[M,N] = act(Abf16[M,lda] @ Wp[K/2,N] + bias)
// Block tile 128x64, K-step 32. 8 waves = 4x2, each wave 2x2 WMMA tiles.
// act: 0=none 1=gelu 2=tanh ; out_bf16: store ushort bf16 instead of f32
// ---------------------------------------------------------------------------
#define GTM 128
#define GTN 64
#define GTK 32

__global__ __launch_bounds__(256) void gemm_wmma_kernel(
    const unsigned short* __restrict__ Ab, long lda,   // elements
    const unsigned int* __restrict__ Wp,               // [K/2,N]
    const float* __restrict__ bias,
    void* __restrict__ Cout, long ldc,
    int M, int N, int K, int act, int out_bf16)
{
  __shared__ unsigned int sA[GTM][20];        // 16 data u32 + pad -> 80B rows (16B aligned)
  __shared__ unsigned int sB[GTK / 2][68];    // 64 data u32 + pad -> 272B rows (16B aligned)

  const int tid  = threadIdx.x;
  const int lane = tid & 31;
  const int wave = tid >> 5;
  const int wm   = wave & 3;
  const int wn   = wave >> 2;
  const int bM   = blockIdx.x * GTM;
  const int bN   = blockIdx.y * GTN;
  const int lm   = lane & 15;
  const int lh   = lane >> 4;

  const unsigned int* Ap = (const unsigned int*)Ab;
  const long ldau = lda >> 1;

  v8f acc[2][2];
  const v8f zero8 = {0.f, 0.f, 0.f, 0.f, 0.f, 0.f, 0.f, 0.f};
#pragma unroll
  for (int i = 0; i < 2; i++)
#pragma unroll
    for (int j = 0; j < 2; j++) acc[i][j] = zero8;

  for (int kt = 0; kt < K; kt += GTK) {
    const int ktp = kt >> 1;
#if USE_ASYNC_LDS
    // A tile: 128 rows x 4 x 16B chunks = 512 lanes-of-b128; 2 per thread
#pragma unroll
    for (int i = 0; i < 2; i++) {
      int idx = tid + i * 256;
      int r = idx >> 2, ch = idx & 3;
      int gr = bM + r; if (gr >= M) gr = M - 1;
      const unsigned int* src = Ap + (long)gr * ldau + ktp + ch * 4;
      unsigned int dst = lds_off(&sA[r][ch * 4]);
      asm volatile("global_load_async_to_lds_b128 %0, %1, off"
                   :: "v"(dst), "v"(src) : "memory");
    }
    // B tile: 16 rows x 16 x 16B chunks = 256 lanes-of-b128; 1 per thread
    {
      int r = tid >> 4, ch = tid & 15;
      const unsigned int* src = Wp + (long)(ktp + r) * N + bN + ch * 4;
      unsigned int dst = lds_off(&sB[r][ch * 4]);
      asm volatile("global_load_async_to_lds_b128 %0, %1, off"
                   :: "v"(dst), "v"(src) : "memory");
    }
    asm volatile("s_wait_asynccnt 0x0" ::: "memory");
#else
#pragma unroll
    for (int i = 0; i < 8; i++) {
      int idx = tid + i * 256;
      int r = idx >> 4, cp = idx & 15;
      int gr = bM + r; if (gr >= M) gr = M - 1;
      sA[r][cp] = Ap[(long)gr * ldau + ktp + cp];
    }
#pragma unroll
    for (int i = 0; i < 4; i++) {
      int idx = tid + i * 256;
      int kp = idx >> 6, c = idx & 63;
      sB[kp][c] = Wp[(long)(ktp + kp) * N + bN + c];
    }
#endif
    __syncthreads();

    BFrag af[2], bf[2];
#pragma unroll
    for (int i = 0; i < 2; i++) {
      int row = wm * 32 + i * 16 + lm;
#pragma unroll
      for (int v = 0; v < 8; v++) af[i].u[v] = sA[row][kmap(v, lh) >> 1];
    }
#pragma unroll
    for (int j = 0; j < 2; j++) {
      int col = wn * 32 + j * 16 + lm;
#pragma unroll
      for (int v = 0; v < 8; v++) bf[j].u[v] = sB[kmap(v, lh) >> 1][col];
    }
#pragma unroll
    for (int i = 0; i < 2; i++)
#pragma unroll
      for (int j = 0; j < 2; j++)
        acc[i][j] = __builtin_amdgcn_wmma_f32_16x16x32_bf16(
            false, af[i].v, false, bf[j].v, (short)0, acc[i][j], false, false);
    __syncthreads();
  }

  // Epilogue: bias + activation + store (C layout: lane n=lm, row r + 8*lh)
#pragma unroll
  for (int i = 0; i < 2; i++)
#pragma unroll
    for (int j = 0; j < 2; j++) {
      int col = bN + wn * 32 + j * 16 + lm;
      float bv = (col < N) ? bias[col] : 0.f;
#pragma unroll
      for (int r = 0; r < 8; r++) {
        int row = bM + wm * 32 + i * 16 + r + lh * 8;
        if (row < M && col < N) {
          float val = acc[i][j][r] + bv;
          if (act == 1)      val = gelu_tanh(val);
          else if (act == 2) val = tanhf(val);
          if (out_bf16) ((unsigned short*)Cout)[(long)row * ldc + col] = bf16_1(val);
          else          ((float*)Cout)[(long)row * ldc + col] = val;
        }
      }
    }
}

// ---------------------------------------------------------------------------
// Attention: one block per (batch, head). Q,K,V are bf16 [B*S, 768].
// scores = (Q K^T)/8 + mask_bias ; softmax ; ctx = P V  (all WMMA bf16)
// Writes ctx directly as bf16.
// ---------------------------------------------------------------------------
__global__ __launch_bounds__(256) void attention_kernel(
    const unsigned short* __restrict__ Qb, const unsigned short* __restrict__ Kb,
    const unsigned short* __restrict__ Vb, const int* __restrict__ amask,
    unsigned short* __restrict__ Ob)
{
  __shared__ unsigned int sQp[128][33];   // Q rows, d-pairs
  __shared__ unsigned int sKp[32][132];   // K d-pairs, per key column
  __shared__ unsigned int sVp[64][68];    // V s-pairs, per d column
  __shared__ float        sS[128][130];   // scores; P (bf16 pairs) overlaid in place

  const int  b   = blockIdx.x;
  const int  h   = blockIdx.y;
  const int  tid = threadIdx.x;
  const long base  = ((long)b * 128) * 768 + h * 64;      // element index
  const long baseu = (long)(b * 128) * 384 + h * 32;      // u32-pair index

  const unsigned int* Qp = (const unsigned int*)Qb;
  const unsigned int* Kp = (const unsigned int*)Kb;

#pragma unroll
  for (int i = 0; i < 16; i++) {
    int idx = tid + i * 256;
    { int r = idx >> 5, dp = idx & 31;
      sQp[r][dp] = Qp[baseu + (long)r * 384 + dp]; }
    { int s = idx & 127, dp = idx >> 7;
      sKp[dp][s] = Kp[baseu + (long)s * 384 + dp]; }
    { int d = idx & 63, sp = idx >> 6;
      unsigned int lo = Vb[base + (long)(2 * sp) * 768 + d];
      unsigned int hi = Vb[base + (long)(2 * sp + 1) * 768 + d];
      sVp[sp][d] = lo | (hi << 16); }
  }
  __syncthreads();

  const int lane = tid & 31, wave = tid >> 5;
  const int wm = wave & 3, wn = wave >> 2;
  const int lm = lane & 15, lh = lane >> 4;
  const v8f zero8 = {0.f, 0.f, 0.f, 0.f, 0.f, 0.f, 0.f, 0.f};

  // ---- scores: wave owns 32 rows x 64 cols (2x4 tiles), K=64 ----
  {
    v8f acc[2][4];
#pragma unroll
    for (int i = 0; i < 2; i++)
#pragma unroll
      for (int j = 0; j < 4; j++) acc[i][j] = zero8;
    for (int kt = 0; kt < 64; kt += 32) {
      BFrag af[2], bf[4];
#pragma unroll
      for (int i = 0; i < 2; i++) {
        int row = wm * 32 + i * 16 + lm;
#pragma unroll
        for (int v = 0; v < 8; v++) af[i].u[v] = sQp[row][(kt + kmap(v, lh)) >> 1];
      }
#pragma unroll
      for (int j = 0; j < 4; j++) {
        int col = wn * 64 + j * 16 + lm;
#pragma unroll
        for (int v = 0; v < 8; v++) bf[j].u[v] = sKp[(kt + kmap(v, lh)) >> 1][col];
      }
#pragma unroll
      for (int i = 0; i < 2; i++)
#pragma unroll
        for (int j = 0; j < 4; j++)
          acc[i][j] = __builtin_amdgcn_wmma_f32_16x16x32_bf16(
              false, af[i].v, false, bf[j].v, (short)0, acc[i][j], false, false);
    }
#pragma unroll
    for (int i = 0; i < 2; i++)
#pragma unroll
      for (int j = 0; j < 4; j++) {
        int col = wn * 64 + j * 16 + lm;
        float mb = (1.0f - (float)amask[b * 128 + col]) * -1.0e9f;
#pragma unroll
        for (int r = 0; r < 8; r++) {
          int row = wm * 32 + i * 16 + r + lh * 8;
          sS[row][col] = acc[i][j][r] * 0.125f + mb;
        }
      }
  }
  __syncthreads();

  // ---- softmax per row; pack P bf16 pairs in place over the score row ----
  if (tid < 128) {
    float mx = -3.4e38f;
    for (int c = 0; c < 128; c++) mx = fmaxf(mx, sS[tid][c]);
    float sm = 0.f;
    for (int c = 0; c < 128; c++) sm += expf(sS[tid][c] - mx);
    float inv = 1.0f / sm;
    unsigned int* prow = (unsigned int*)&sS[tid][0];
    for (int cp = 0; cp < 64; cp++) {     // writes idx cp after reading 2cp,2cp+1: safe
      float e0 = expf(sS[tid][2 * cp] - mx) * inv;
      float e1 = expf(sS[tid][2 * cp + 1] - mx) * inv;
      prow[cp] = pack_bf16(e0, e1);
    }
  }
  __syncthreads();

  // ---- ctx = P(128x128) V(128x64): wave owns 32x32 (2x2 tiles), K=128 ----
  {
    v8f acc[2][2];
#pragma unroll
    for (int i = 0; i < 2; i++)
#pragma unroll
      for (int j = 0; j < 2; j++) acc[i][j] = zero8;
    for (int kt = 0; kt < 128; kt += 32) {
      BFrag af[2], bf[2];
#pragma unroll
      for (int i = 0; i < 2; i++) {
        int row = wm * 32 + i * 16 + lm;
        const unsigned int* prow = (const unsigned int*)&sS[row][0];
#pragma unroll
        for (int v = 0; v < 8; v++) af[i].u[v] = prow[(kt + kmap(v, lh)) >> 1];
      }
#pragma unroll
      for (int j = 0; j < 2; j++) {
        int col = wn * 32 + j * 16 + lm;
#pragma unroll
        for (int v = 0; v < 8; v++) bf[j].u[v] = sVp[(kt + kmap(v, lh)) >> 1][col];
      }
#pragma unroll
      for (int i = 0; i < 2; i++)
#pragma unroll
        for (int j = 0; j < 2; j++)
          acc[i][j] = __builtin_amdgcn_wmma_f32_16x16x32_bf16(
              false, af[i].v, false, bf[j].v, (short)0, acc[i][j], false, false);
    }
#pragma unroll
    for (int i = 0; i < 2; i++)
#pragma unroll
      for (int j = 0; j < 2; j++) {
        int col = wn * 32 + j * 16 + lm;
#pragma unroll
        for (int r = 0; r < 8; r++) {
          int row = wm * 32 + i * 16 + r + lh * 8;
          Ob[base + (long)row * 768 + col] = bf16_1(acc[i][j][r]);
        }
      }
  }
}

// ---------------------------------------------------------------------------
// Embedding + LayerNorm: one block/token. Writes f32 (residual) + bf16 (GEMM A).
// ---------------------------------------------------------------------------
__global__ __launch_bounds__(256) void embed_ln_kernel(
    const int* __restrict__ ids, const float* __restrict__ wemb,
    const float* __restrict__ pemb, const float* __restrict__ temb,
    const float* __restrict__ lnw, const float* __restrict__ lnb,
    float* __restrict__ x, unsigned short* __restrict__ xb)
{
  __shared__ float red[256];
  int row = blockIdx.x;
  int s   = row & 127;
  int id  = ids[row];
  float vals[3], lsum = 0.f;
#pragma unroll
  for (int i = 0; i < 3; i++) {
    int c = threadIdx.x + i * 256;
    float v = wemb[(long)id * 768 + c] + pemb[(long)s * 768 + c] + temb[c];
    vals[i] = v; lsum += v;
  }
  red[threadIdx.x] = lsum; __syncthreads();
  for (int t = 128; t > 0; t >>= 1) { if (threadIdx.x < t) red[threadIdx.x] += red[threadIdx.x + t]; __syncthreads(); }
  float mean = red[0] * (1.0f / 768.f); __syncthreads();
  float lvar = 0.f;
#pragma unroll
  for (int i = 0; i < 3; i++) { float d = vals[i] - mean; lvar += d * d; }
  red[threadIdx.x] = lvar; __syncthreads();
  for (int t = 128; t > 0; t >>= 1) { if (threadIdx.x < t) red[threadIdx.x] += red[threadIdx.x + t]; __syncthreads(); }
  float rstd = rsqrtf(red[0] * (1.0f / 768.f) + 1e-12f);
#pragma unroll
  for (int i = 0; i < 3; i++) {
    int c = threadIdx.x + i * 256;
    float o = (vals[i] - mean) * rstd * lnw[c] + lnb[c];
    x[(long)row * 768 + c]  = o;
    xb[(long)row * 768 + c] = bf16_1(o);
  }
}

// out = LayerNorm(x + delta); writes f32 + bf16. In-place over x is safe.
__global__ __launch_bounds__(256) void add_ln_kernel(
    const float* __restrict__ x, const float* __restrict__ delta,
    const float* __restrict__ lnw, const float* __restrict__ lnb,
    float* __restrict__ out, unsigned short* __restrict__ outb)
{
  __shared__ float red[256];
  int row = blockIdx.x;
  float vals[3], lsum = 0.f;
#pragma unroll
  for (int i = 0; i < 3; i++) {
    int c = threadIdx.x + i * 256;
    float v = x[(long)row * 768 + c] + delta[(long)row * 768 + c];
    vals[i] = v; lsum += v;
  }
  red[threadIdx.x] = lsum; __syncthreads();
  for (int t = 128; t > 0; t >>= 1) { if (threadIdx.x < t) red[threadIdx.x] += red[threadIdx.x + t]; __syncthreads(); }
  float mean = red[0] * (1.0f / 768.f); __syncthreads();
  float lvar = 0.f;
#pragma unroll
  for (int i = 0; i < 3; i++) { float d = vals[i] - mean; lvar += d * d; }
  red[threadIdx.x] = lvar; __syncthreads();
  for (int t = 128; t > 0; t >>= 1) { if (threadIdx.x < t) red[threadIdx.x] += red[threadIdx.x + t]; __syncthreads(); }
  float rstd = rsqrtf(red[0] * (1.0f / 768.f) + 1e-12f);
#pragma unroll
  for (int i = 0; i < 3; i++) {
    int c = threadIdx.x + i * 256;
    float o = (vals[i] - mean) * rstd * lnw[c] + lnb[c];
    out[(long)row * 768 + c]  = o;
    outb[(long)row * 768 + c] = bf16_1(o);
  }
}

// ---------------------------------------------------------------------------
// Tiny heads + CRF (unchanged from v1)
// ---------------------------------------------------------------------------
__global__ void ner_logits_kernel(const float* __restrict__ x,
                                  const float* __restrict__ w,
                                  const float* __restrict__ bvec,
                                  float* __restrict__ out)
{
  int idx = blockIdx.x * blockDim.x + threadIdx.x;
  if (idx >= 8192 * 5) return;
  int row = idx / 5, t = idx % 5;
  const float* xr = x + (long)row * 768;
  float a = bvec[t];
  for (int k = 0; k < 768; k++) a += xr[k] * w[k * 5 + t];
  out[idx] = a;
}

__global__ void cls_head_kernel(const float* __restrict__ pooled,
                                const float* __restrict__ w,
                                const float* __restrict__ bvec,
                                const int* __restrict__ labels,
                                const float* __restrict__ cw,
                                float* __restrict__ out)
{
  __shared__ float rn[64], rd[64];
  int b = threadIdx.x;
  const float* pr = pooled + (long)b * 768;
  float lg[5];
#pragma unroll
  for (int t = 0; t < 5; t++) {
    float a = bvec[t];
    for (int k = 0; k < 768; k++) a += pr[k] * w[k * 5 + t];
    lg[t] = a;
    out[2 + b * 5 + t] = a;
  }
  float mx = lg[0];
#pragma unroll
  for (int t = 1; t < 5; t++) mx = fmaxf(mx, lg[t]);
  float sm = 0.f;
#pragma unroll
  for (int t = 0; t < 5; t++) sm += expf(lg[t] - mx);
  float lse = mx + logf(sm);
  int lab = labels[b];
  float wv = cw[lab];
  rn[b] = wv * (lse - lg[lab]);
  rd[b] = wv;
  __syncthreads();
  if (b == 0) {
    float n = 0.f, d = 0.f;
    for (int i = 0; i < 64; i++) { n += rn[i]; d += rd[i]; }
    out[0] = n / d;
  }
}

__global__ void crf_loss_kernel(const float* __restrict__ em_all,
                                const int* __restrict__ tags_all,
                                const int* __restrict__ mask_all,
                                const float* __restrict__ startv,
                                const float* __restrict__ endv,
                                const float* __restrict__ trans,
                                float* __restrict__ out)
{
  __shared__ float red[64];
  int b = threadIdx.x;
  const float* em  = em_all + (long)b * 128 * 5;
  const int* tags  = tags_all + b * 128;
  const int* mask  = mask_all + b * 128;

  int msum = 0;
  for (int s = 0; s < 128; s++) msum += mask[s];
  int seq_end = msum - 1;

  float numer = startv[tags[0]] + em[tags[0]];
  int prev = tags[0];
  for (int s = 1; s < 128; s++) {
    float m = (float)mask[s];
    int tg = tags[s];
    numer += (em[s * 5 + tg] + trans[prev * 5 + tg]) * m;
    prev = tg;
  }
  numer += endv[tags[seq_end]];

  float alpha[5];
#pragma unroll
  for (int j = 0; j < 5; j++) alpha[j] = startv[j] + em[j];
  for (int s = 1; s < 128; s++) {
    float nxt[5];
#pragma unroll
    for (int j = 0; j < 5; j++) {
      float mx = -3.4e38f;
#pragma unroll
      for (int i = 0; i < 5; i++) mx = fmaxf(mx, alpha[i] + trans[i * 5 + j]);
      float sm = 0.f;
#pragma unroll
      for (int i = 0; i < 5; i++) sm += expf(alpha[i] + trans[i * 5 + j] - mx);
      nxt[j] = mx + logf(sm) + em[s * 5 + j];
    }
    if (mask[s]) {
#pragma unroll
      for (int j = 0; j < 5; j++) alpha[j] = nxt[j];
    }
  }
  float mx = -3.4e38f;
#pragma unroll
  for (int j = 0; j < 5; j++) mx = fmaxf(mx, alpha[j] + endv[j]);
  float sm = 0.f;
#pragma unroll
  for (int j = 0; j < 5; j++) sm += expf(alpha[j] + endv[j] - mx);
  float denom = mx + logf(sm);

  red[b] = -(numer - denom);
  __syncthreads();
  if (b == 0) {
    float acc = 0.f;
    for (int i = 0; i < 64; i++) acc += red[i];
    out[1] = acc * (1.0f / 64.f);
  }
}

__global__ void viterbi_kernel(const float* __restrict__ em_all,
                               const int* __restrict__ mask_all,
                               const float* __restrict__ startv,
                               const float* __restrict__ endv,
                               const float* __restrict__ trans,
                               int* __restrict__ hist_ws,
                               float* __restrict__ out_tags)
{
  int b = threadIdx.x;
  const float* em = em_all + (long)b * 128 * 5;
  const int* mask = mask_all + b * 128;
  int* hist = hist_ws + (long)b * 127 * 5;

  float score[5];
#pragma unroll
  for (int j = 0; j < 5; j++) score[j] = startv[j] + em[j];

  for (int s = 1; s < 128; s++) {
    float nxt[5]; int bp[5];
#pragma unroll
    for (int j = 0; j < 5; j++) {
      float best = score[0] + trans[j];
      int bi = 0;
#pragma unroll
      for (int i = 1; i < 5; i++) {
        float c = score[i] + trans[i * 5 + j];
        if (c > best) { best = c; bi = i; }
      }
      nxt[j] = best + em[s * 5 + j];
      bp[j] = bi;
    }
    if (mask[s]) {
#pragma unroll
      for (int j = 0; j < 5; j++) { score[j] = nxt[j]; hist[(s - 1) * 5 + j] = bp[j]; }
    } else {
#pragma unroll
      for (int j = 0; j < 5; j++) hist[(s - 1) * 5 + j] = j;
    }
  }
  int last = 0;
  float bestf = score[0] + endv[0];
#pragma unroll
  for (int j = 1; j < 5; j++) {
    float c = score[j] + endv[j];
    if (c > bestf) { bestf = c; last = j; }
  }
  out_tags[b * 128 + 127] = (float)last;
  int tag = last;
  for (int s = 127; s >= 1; s--) {
    tag = hist[(s - 1) * 5 + tag];
    out_tags[b * 128 + s - 1] = (float)tag;
  }
}

// ---------------------------------------------------------------------------
extern "C" void kernel_launch(void* const* d_in, const int* in_sizes, int n_in,
                              void* d_out, int out_size, void* d_ws, size_t ws_size,
                              hipStream_t stream) {
  (void)in_sizes; (void)n_in; (void)out_size; (void)ws_size;

  const int*   ids   = (const int*)d_in[0];
  const int*   amask = (const int*)d_in[1];
  const int*   nmask = (const int*)d_in[2];
  const int*   acs   = (const int*)d_in[3];
  const int*   ntags = (const int*)d_in[4];
  const float* cw    = (const float*)d_in[5];
  const float* wemb  = (const float*)d_in[6];
  const float* pemb  = (const float*)d_in[7];
  const float* temb  = (const float*)d_in[8];
  const float* elnw  = (const float*)d_in[9];
  const float* elnb  = (const float*)d_in[10];
  const float* WqA   = (const float*)d_in[11];
  const float* bqA   = (const float*)d_in[12];
  const float* WkA   = (const float*)d_in[13];
  const float* bkA   = (const float*)d_in[14];
  const float* WvA   = (const float*)d_in[15];
  const float* bvA   = (const float*)d_in[16];
  const float* WoA   = (const float*)d_in[17];
  const float* boA   = (const float*)d_in[18];
  const float* l1wA  = (const float*)d_in[19];
  const float* l1bA  = (const float*)d_in[20];
  const float* W1A   = (const float*)d_in[21];
  const float* b1A   = (const float*)d_in[22];
  const float* W2A   = (const float*)d_in[23];
  const float* b2A   = (const float*)d_in[24];
  const float* l2wA  = (const float*)d_in[25];
  const float* l2bA  = (const float*)d_in[26];
  const float* poolw = (const float*)d_in[27];
  const float* poolb = (const float*)d_in[28];
  const float* clsw  = (const float*)d_in[29];
  const float* clsb  = (const float*)d_in[30];
  const float* nerw  = (const float*)d_in[31];
  const float* nerb  = (const float*)d_in[32];
  const float* crfs  = (const float*)d_in[33];
  const float* crfe  = (const float*)d_in[34];
  const float* crft  = (const float*)d_in[35];

  // Workspace layout. FF bf16 buffer overlays dead Q/K/V/CTX bf16 exactly.
  const size_t SZ_F32 = (size_t)8192 * 768 * 4;   // 25165824
  const size_t SZ_BF  = (size_t)8192 * 768 * 2;   // 12582912
  char* ws = (char*)d_ws;
  size_t off = 0;
  float* X    = (float*)(ws + off); off += SZ_F32;
  float* TMP  = (float*)(ws + off); off += SZ_F32;
  unsigned short* Xb   = (unsigned short*)(ws + off); off += SZ_BF;
  unsigned short* Qb   = (unsigned short*)(ws + off); off += SZ_BF;
  unsigned short* Kb   = (unsigned short*)(ws + off); off += SZ_BF;
  unsigned short* Vb   = (unsigned short*)(ws + off); off += SZ_BF;
  unsigned short* CTXb = (unsigned short*)(ws + off); off += SZ_BF;
  unsigned short* FFb  = Qb;                      // 4*SZ_BF == 8192*3072*2
  float* NERL = (float*)(ws + off); off += (size_t)8192 * 5 * 4;
  float* POOL = (float*)(ws + off); off += (size_t)64 * 768 * 4;
  int*   HIST = (int*)(ws + off);   off += (size_t)64 * 127 * 5 * 4;
  unsigned int* WSq = (unsigned int*)(ws + off); off += (size_t)384 * 768 * 4;
  unsigned int* WSk = (unsigned int*)(ws + off); off += (size_t)384 * 768 * 4;
  unsigned int* WSv = (unsigned int*)(ws + off); off += (size_t)384 * 768 * 4;
  unsigned int* WSo = (unsigned int*)(ws + off); off += (size_t)384 * 768 * 4;
  unsigned int* WS1 = (unsigned int*)(ws + off); off += (size_t)384 * 3072 * 4;
  unsigned int* WS2 = (unsigned int*)(ws + off); off += (size_t)1536 * 768 * 4;
  unsigned int* WSp = (unsigned int*)(ws + off); off += (size_t)384 * 768 * 4;
  float* OUT = (float*)d_out;

  embed_ln_kernel<<<8192, 256, 0, stream>>>(ids, wemb, pemb, temb, elnw, elnb, X, Xb);

  for (int l = 0; l < 12; l++) {
    const float* Wq = WqA + (long)l * 768 * 768;
    const float* Wk = WkA + (long)l * 768 * 768;
    const float* Wv = WvA + (long)l * 768 * 768;
    const float* Wo = WoA + (long)l * 768 * 768;
    const float* W1 = W1A + (long)l * 768 * 3072;
    const float* W2 = W2A + (long)l * 3072 * 768;
    const float* bq = bqA + l * 768, *bk = bkA + l * 768, *bv = bvA + l * 768;
    const float* bo = boA + l * 768, *b1 = b1A + l * 3072, *b2 = b2A + l * 768;
    const float* l1w = l1wA + l * 768, *l1b = l1bA + l * 768;
    const float* l2w = l2wA + l * 768, *l2b = l2bA + l * 768;

    // stage this layer's weights in bf16 pair-along-K layout
    convert_w_kernel<<<1152, 256, 0, stream>>>(Wq, WSq, 384, 768);
    convert_w_kernel<<<1152, 256, 0, stream>>>(Wk, WSk, 384, 768);
    convert_w_kernel<<<1152, 256, 0, stream>>>(Wv, WSv, 384, 768);
    convert_w_kernel<<<1152, 256, 0, stream>>>(Wo, WSo, 384, 768);
    convert_w_kernel<<<4608, 256, 0, stream>>>(W1, WS1, 384, 3072);
    convert_w_kernel<<<4608, 256, 0, stream>>>(W2, WS2, 1536, 768);

    gemm_wmma_kernel<<<dim3(64, 12), 256, 0, stream>>>(Xb, 768, WSq, bq, Qb, 768, 8192, 768, 768, 0, 1);
    gemm_wmma_kernel<<<dim3(64, 12), 256, 0, stream>>>(Xb, 768, WSk, bk, Kb, 768, 8192, 768, 768, 0, 1);
    gemm_wmma_kernel<<<dim3(64, 12), 256, 0, stream>>>(Xb, 768, WSv, bv, Vb, 768, 8192, 768, 768, 0, 1);
    attention_kernel<<<dim3(64, 12), 256, 0, stream>>>(Qb, Kb, Vb, amask, CTXb);
    gemm_wmma_kernel<<<dim3(64, 12), 256, 0, stream>>>(CTXb, 768, WSo, bo, TMP, 768, 8192, 768, 768, 0, 0);
    add_ln_kernel<<<8192, 256, 0, stream>>>(X, TMP, l1w, l1b, X, Xb);
    gemm_wmma_kernel<<<dim3(64, 48), 256, 0, stream>>>(Xb, 768, WS1, b1, FFb, 3072, 8192, 3072, 768, 1, 1);
    gemm_wmma_kernel<<<dim3(64, 12), 256, 0, stream>>>(FFb, 3072, WS2, b2, TMP, 768, 8192, 768, 3072, 0, 0);
    add_ln_kernel<<<8192, 256, 0, stream>>>(X, TMP, l2w, l2b, X, Xb);
  }

  // pooled = tanh(x[:,0] @ pool_w + pool_b): CLS rows strided by 128*768
  convert_w_kernel<<<1152, 256, 0, stream>>>(poolw, WSp, 384, 768);
  gemm_wmma_kernel<<<dim3(1, 12), 256, 0, stream>>>(Xb, 128 * 768, WSp, poolb, POOL, 768, 64, 768, 768, 2, 0);

  ner_logits_kernel<<<160, 256, 0, stream>>>(X, nerw, nerb, NERL);
  cls_head_kernel<<<1, 64, 0, stream>>>(POOL, clsw, clsb, acs, cw, OUT);
  crf_loss_kernel<<<1, 64, 0, stream>>>(NERL, ntags, nmask, crfs, crfe, crft, OUT);
  viterbi_kernel<<<1, 64, 0, stream>>>(NERL, nmask, crfs, crfe, crft, HIST, OUT + 322);
}
